// GatedDeltaNetLayer_78357383348349
// MI455X (gfx1250) — matrix operand order repaired and checked by reading, not compiled, not run on hardware
//
#include <hip/hip_runtime.h>
#include <hip/hip_bf16.h>
#include <math.h>
#include <stdint.h>

// ---------------------------------------------------------------------------
// Gated DeltaNet layer for gfx1250 (MI455X).
// GEMMs run on v_wmma_f32_16x16x32_bf16; the sequential delta-rule scan keeps
// fp32 state in registers (one V-column per lane, columns are independent) and
// software-pipelines its k-row fetch with global_load_async_to_lds_b128.
// ---------------------------------------------------------------------------

typedef __bf16 bf16_t;
typedef __attribute__((ext_vector_type(4)))  __bf16 v4bf;
typedef __attribute__((ext_vector_type(8)))  __bf16 v8bf;
typedef __attribute__((ext_vector_type(16))) __bf16 v16bf;
typedef __attribute__((ext_vector_type(4)))  float  v4f;
typedef __attribute__((ext_vector_type(8)))  float  v8f;

#define B_    4
#define T_    4096
#define D_    1024
#define H_    16
#define K_    64
#define V_    64
#define NROWS (B_ * T_)   // 16384
#define NPROJ 2112        // 1024(k) + 1024(v) + 16(beta) + 16(g) + 32 pad, = 33*64
#define EPSF  1e-6f

// --------------------------------------------------------------------------
// Kernel 1: RMSNorm, fp32 -> bf16.  One block (256 threads) per row of 1024.
// --------------------------------------------------------------------------
__global__ __launch_bounds__(256)
void rmsnorm_kernel(const float* __restrict__ x, const float* __restrict__ w,
                    bf16_t* __restrict__ xn) {
    __shared__ float red[256];
    const int row = blockIdx.x;
    const int tid = threadIdx.x;
    const float* xr = x + (size_t)row * D_;

    v4f v = *(const v4f*)(xr + tid * 4);
    red[tid] = v.x * v.x + v.y * v.y + v.z * v.z + v.w * v.w;
    __syncthreads();
    for (int off = 128; off > 0; off >>= 1) {
        if (tid < off) red[tid] += red[tid + off];
        __syncthreads();
    }
    const float inv = 1.0f / sqrtf(red[0] * (1.0f / (float)D_) + EPSF);

    v4f wv = *(const v4f*)(w + tid * 4);
    v4bf o;
    o[0] = (bf16_t)(v.x * inv * wv.x);
    o[1] = (bf16_t)(v.y * inv * wv.y);
    o[2] = (bf16_t)(v.z * inv * wv.z);
    o[3] = (bf16_t)(v.w * inv * wv.w);
    *(v4bf*)(xn + (size_t)row * D_ + tid * 4) = o;
}

// --------------------------------------------------------------------------
// Kernel 2: pack [Wk; Wv; Wb; Wg] (+zero pad) into bf16 [NPROJ, D], and
// convert Wo -> bf16 [D, H*V].
// --------------------------------------------------------------------------
__global__ __launch_bounds__(256)
void pack_weights_kernel(const float* __restrict__ Wk, const float* __restrict__ Wv,
                         const float* __restrict__ Wb, const float* __restrict__ Wg,
                         const float* __restrict__ Wo,
                         bf16_t* __restrict__ Wcat, bf16_t* __restrict__ Wob) {
    const int idx = blockIdx.x * 256 + threadIdx.x;
    const int total1 = NPROJ * D_;
    if (idx < total1) {
        const int r = idx / D_;
        const int c = idx - r * D_;
        float val = 0.0f;
        if (r < 1024)      val = Wk[(size_t)r * D_ + c];
        else if (r < 2048) val = Wv[(size_t)(r - 1024) * D_ + c];
        else if (r < 2064) val = Wb[(size_t)(r - 2048) * D_ + c];
        else if (r < 2080) val = Wg[(size_t)(r - 2064) * D_ + c];
        Wcat[idx] = (bf16_t)val;
    } else {
        const int j = idx - total1;
        if (j < D_ * D_) Wob[j] = (bf16_t)Wo[j];
    }
}

// --------------------------------------------------------------------------
// WMMA fragment loaders (contiguous k order used by this code for both A and B, not the ISA table order; 16-bit A 16x32 and
// 16-bit B 32x16, wave32).
//   A (row-major [M,K]): lane L -> m = L&15; elems 0..15 = K[16*hi .. +15],
//                        contiguous, hi = L>>4.
//   B stored as B^T, row-major [N,K]: lane L -> n = L&15;
//                        elems 0..15 = K[16*hi .. 16*hi+15] contiguous.
// --------------------------------------------------------------------------
__device__ inline v16bf load_a_frag(const bf16_t* __restrict__ A, int lda,
                                    int mBase, int kBase, int lane) {
    const int m  = mBase + (lane & 15);
    const int hi = lane >> 4;
    const bf16_t* p = A + (size_t)m * lda + kBase + 16 * hi;
    v8bf lo = *(const v8bf*)p;
    v8bf up = *(const v8bf*)(p + 8);
    v16bf r;
#pragma unroll
    for (int i = 0; i < 8; ++i) { r[i] = lo[i]; r[i + 8] = up[i]; }
    return r;
}

__device__ inline v16bf load_b_frag(const bf16_t* __restrict__ Bt, int ldb,
                                    int nBase, int kBase, int lane) {
    const int n  = nBase + (lane & 15);
    const int ko = kBase + 16 * (lane >> 4);
    return *(const v16bf*)(Bt + (size_t)n * ldb + ko);
}

// --------------------------------------------------------------------------
// Kernel 3/5: bf16 WMMA GEMM, C = A * B^T (+ optional fp32 residual).
// One wave per 64x64 output tile: 4 A-frags x 4 B-frags -> 16 WMMAs / k-step.
// If Cb != null: store bf16 to Cb; else store fp32 (acc + resid) to Cf.
// --------------------------------------------------------------------------
__global__ __launch_bounds__(32)
void gemm_bf16_wmma_kernel(const bf16_t* __restrict__ A, int lda,
                           const bf16_t* __restrict__ Bt, int ldb, int Kd,
                           bf16_t* __restrict__ Cb, float* __restrict__ Cf,
                           const float* __restrict__ resid, int ldc) {
    const int lane  = threadIdx.x & 31;
    const int mBase = blockIdx.x * 64;
    const int nBase = blockIdx.y * 64;

    v8f acc[4][4];
#pragma unroll
    for (int i = 0; i < 4; ++i)
#pragma unroll
        for (int j = 0; j < 4; ++j) acc[i][j] = (v8f){0.f,0.f,0.f,0.f,0.f,0.f,0.f,0.f};

    for (int k = 0; k < Kd; k += 32) {
        v16bf af[4], bfrag[4];
#pragma unroll
        for (int i = 0; i < 4; ++i) af[i]    = load_a_frag(A, lda, mBase + 16 * i, k, lane);
#pragma unroll
        for (int j = 0; j < 4; ++j) bfrag[j] = load_b_frag(Bt, ldb, nBase + 16 * j, k, lane);
#pragma unroll
        for (int i = 0; i < 4; ++i)
#pragma unroll
            for (int j = 0; j < 4; ++j)
                acc[i][j] = __builtin_amdgcn_wmma_f32_16x16x32_bf16(
                    false, af[i], false, bfrag[j], (short)0, acc[i][j], false, false);
    }

    // C/D layout: VGPR r, lane L -> m = r + 8*(L>>4), n = L&15.
    const int nOff = lane & 15;
    const int mOff = 8 * (lane >> 4);
#pragma unroll
    for (int i = 0; i < 4; ++i) {
#pragma unroll
        for (int j = 0; j < 4; ++j) {
#pragma unroll
            for (int r = 0; r < 8; ++r) {
                const int m = mBase + 16 * i + mOff + r;
                const int n = nBase + 16 * j + nOff;
                const size_t idx = (size_t)m * ldc + n;
                const float vv = acc[i][j][r];
                if (Cb) Cb[idx] = (bf16_t)vv;
                else    Cf[idx] = vv + resid[idx];
            }
        }
    }
}

// --------------------------------------------------------------------------
// Kernel 4: gated delta-rule scan. One block per (b,h); thread j owns state
// column j (64 fp32 regs). The k-row for step t+1 is DMA'd into a
// double-buffered LDS slot with global_load_async_to_lds_b128 while step t
// computes; completion enforced with s_wait_asynccnt + barrier.
// --------------------------------------------------------------------------
__device__ inline void async_k_prefetch(const bf16_t* gsrc, bf16_t* ldst) {
    unsigned long long g = (unsigned long long)(uintptr_t)gsrc;
    unsigned int       l = (unsigned int)(uintptr_t)(void*)ldst;  // flat LDS: low 32b = LDS addr
    asm volatile("global_load_async_to_lds_b128 %0, %1, off" :: "v"(l), "v"(g) : "memory");
}

__global__ __launch_bounds__(64)
void scan_kernel(const bf16_t* __restrict__ proj,
                 const float* __restrict__ bb, const float* __restrict__ bg,
                 bf16_t* __restrict__ outbuf) {
    const int bh = blockIdx.x;
    const int b  = bh >> 4;
    const int h  = bh & 15;
    const int j  = threadIdx.x;   // 0..63  (V column)

    __shared__ __align__(16) bf16_t kbuf[2][K_];   // double-buffered bf16 k-row

    float st[K_];
#pragma unroll
    for (int i = 0; i < K_; ++i) st[i] = 0.0f;

    const float bbh = bb[h];
    const float bgh = bg[h];
    const bf16_t* base = proj + (size_t)b * T_ * NPROJ;

    // Preload k-row for t=0 into buffer 0 (lanes 0..7, 16B each).
    if (j < 8) async_k_prefetch(base + h * K_ + j * 8, &kbuf[0][j * 8]);
    asm volatile("s_wait_asynccnt 0x0" ::: "memory");
    __syncthreads();

    int cur = 0;
    for (int t = 0; t < T_; ++t) {
        const bf16_t* pr = base + (size_t)t * NPROJ;

        // Kick off async DMA of the next k-row into the other buffer.
        const int tn = (t + 1 < T_) ? (t + 1) : t;
        if (j < 8)
            async_k_prefetch(base + (size_t)tn * NPROJ + h * K_ + j * 8,
                             &kbuf[cur ^ 1][j * 8]);
        // Counter-free prefetch of the v/beta/g lines a few steps ahead.
        if (t + 4 < T_)
            __builtin_prefetch(pr + 4 * (size_t)NPROJ + 1024 + h * V_ + j, 0, 0);

        // Unpack k (8x ds_load_b128) and compute its L2 norm.
        float kr[K_];
        float s0 = 0.f, s1 = 0.f, s2 = 0.f, s3 = 0.f;
#pragma unroll
        for (int i = 0; i < K_; i += 8) {
            v8bf kb = *(const v8bf*)&kbuf[cur][i];
#pragma unroll
            for (int u = 0; u < 8; ++u) kr[i + u] = (float)kb[u];
            s0 = fmaf(kr[i],     kr[i],     s0);
            s1 = fmaf(kr[i + 1], kr[i + 1], s1);
            s2 = fmaf(kr[i + 2], kr[i + 2], s2);
            s3 = fmaf(kr[i + 3], kr[i + 3], s3);
            s0 = fmaf(kr[i + 4], kr[i + 4], s0);
            s1 = fmaf(kr[i + 5], kr[i + 5], s1);
            s2 = fmaf(kr[i + 6], kr[i + 6], s2);
            s3 = fmaf(kr[i + 7], kr[i + 7], s3);
        }
        const float inv = 1.0f / fmaxf(sqrtf(s0 + s1 + s2 + s3), 1e-12f);
#pragma unroll
        for (int i = 0; i < K_; ++i) kr[i] *= inv;

        // pred = sum_i st[i] * kr[i]
        float p0 = 0.f, p1 = 0.f, p2 = 0.f, p3 = 0.f;
#pragma unroll
        for (int i = 0; i < K_; i += 4) {
            p0 = fmaf(st[i],     kr[i],     p0);
            p1 = fmaf(st[i + 1], kr[i + 1], p1);
            p2 = fmaf(st[i + 2], kr[i + 2], p2);
            p3 = fmaf(st[i + 3], kr[i + 3], p3);
        }
        const float vj   = (float)pr[1024 + h * V_ + j];
        const float err  = vj - (p0 + p1 + p2 + p3);
        const float beta = 1.0f / (1.0f + __expf(-((float)pr[2048 + h] + bbh)));
        const float g    = 1.0f / (1.0f + __expf(-((float)pr[2064 + h] + bgh)));
        const float c    = beta * err;

        // st = g*st + c*kr ;  out = sum_i st[i]*kr[i]
        float o0 = 0.f, o1 = 0.f, o2 = 0.f, o3 = 0.f;
#pragma unroll
        for (int i = 0; i < K_; i += 4) {
            st[i]     = fmaf(c, kr[i],     g * st[i]);
            st[i + 1] = fmaf(c, kr[i + 1], g * st[i + 1]);
            st[i + 2] = fmaf(c, kr[i + 2], g * st[i + 2]);
            st[i + 3] = fmaf(c, kr[i + 3], g * st[i + 3]);
            o0 = fmaf(st[i],     kr[i],     o0);
            o1 = fmaf(st[i + 1], kr[i + 1], o1);
            o2 = fmaf(st[i + 2], kr[i + 2], o2);
            o3 = fmaf(st[i + 3], kr[i + 3], o3);
        }
        outbuf[((size_t)b * T_ + t) * (H_ * V_) + h * V_ + j] =
            (bf16_t)(o0 + o1 + o2 + o3);

        // Next k-row must have landed in LDS before the next iteration.
        asm volatile("s_wait_asynccnt 0x0" ::: "memory");
        __syncthreads();
        cur ^= 1;
    }
}

// --------------------------------------------------------------------------
// Launcher
// --------------------------------------------------------------------------
extern "C" void kernel_launch(void* const* d_in, const int* in_sizes, int n_in,
                              void* d_out, int out_size, void* d_ws, size_t ws_size,
                              hipStream_t stream) {
    const float* x  = (const float*)d_in[0];
    const float* nw = (const float*)d_in[1];
    const float* Wk = (const float*)d_in[2];
    const float* Wv = (const float*)d_in[3];
    const float* Wo = (const float*)d_in[4];
    const float* Wb = (const float*)d_in[5];
    const float* bb = (const float*)d_in[6];
    const float* Wg = (const float*)d_in[7];
    const float* bg = (const float*)d_in[8];
    float* y = (float*)d_out;

    // Workspace layout (all offsets are multiples of 32B; total ~143 MB).
    char* w = (char*)d_ws;
    bf16_t* xn   = (bf16_t*)w;  w += (size_t)NROWS * D_ * sizeof(bf16_t);
    bf16_t* Wcat = (bf16_t*)w;  w += (size_t)NPROJ * D_ * sizeof(bf16_t);
    bf16_t* Wob  = (bf16_t*)w;  w += (size_t)D_ * D_ * sizeof(bf16_t);
    bf16_t* proj = (bf16_t*)w;  w += (size_t)NROWS * NPROJ * sizeof(bf16_t);
    bf16_t* obuf = (bf16_t*)w;

    // 1) RMSNorm -> bf16
    rmsnorm_kernel<<<NROWS, 256, 0, stream>>>(x, nw, xn);

    // 2) Pack weights -> bf16
    const int packTotal = NPROJ * D_ + D_ * D_;
    pack_weights_kernel<<<(packTotal + 255) / 256, 256, 0, stream>>>(
        Wk, Wv, Wb, Wg, Wo, Wcat, Wob);

    // 3) Fused projection GEMM: proj[16384, 2112] = xn @ Wcat^T   (WMMA bf16)
    dim3 g1(NROWS / 64, NPROJ / 64);
    gemm_bf16_wmma_kernel<<<g1, 32, 0, stream>>>(
        xn, D_, Wcat, D_, D_, proj, nullptr, nullptr, NPROJ);

    // 4) Gated delta-rule scan: obuf[16384, 1024]
    scan_kernel<<<B_ * H_, 64, 0, stream>>>(proj, bb, bg, obuf);

    // 5) Output GEMM with fused residual: y = x + obuf @ Wo^T   (WMMA bf16)
    dim3 g2(NROWS / 64, D_ / 64);
    gemm_bf16_wmma_kernel<<<g2, 32, 0, stream>>>(
        obuf, H_ * V_, Wob, H_ * V_, H_ * V_, nullptr, y, x, D_);
}
